// GnnRnnAgent_910533067584
// MI455X (gfx1250) — matrix operand order repaired and checked by reading, not compiled, not run on hardware
//
#include <hip/hip_runtime.h>

// ---------------- problem constants ----------------
#define B_EP      4096
#define A_AG      8
#define N_NODES   (B_EP * A_AG)        // 32768
#define NEPW      4                    // episodes per workgroup
#define NODES     32                   // nodes per workgroup (= 2 WMMA M-tiles)
#define NTHREADS  128                  // 4 waves of 32

typedef __attribute__((ext_vector_type(16))) __bf16 v16bf;
typedef __attribute__((ext_vector_type(8)))  float  v8f;

union FragU { unsigned int u[8]; v16bf b; };

// ---------------- LDS layout (bytes) ----------------
// OFF_W   : bf16 [512][72]  stage-1 weights  -> reused as bf16 [384][72] GRU weights
// OFF_X   : bf16 [ 32][68]  x tile
// OFF_XLR : f32  [ 32][520] xl|xr            -> reused as f32 [32][400] gi|gh
// OFF_HH  : bf16 [ 32][136] h_gnn | hidden
// OFF_AGG : f32  [ 32][256] per-head agg     (stage 0: async-staged fp32 x tile)
//                                            (stage 4+: f32 [32][64] h + LN stats)
#define OFF_W      0
#define OFF_X      73728
#define OFF_XLR    78080
#define OFF_HH     144640
#define OFF_AGG    153344
#define OFF_HOUT   153344
#define OFF_STAT   161536
#define SMEM_BYTES 186112

#define WMMA_BF16(A, B, C) \
  __builtin_amdgcn_wmma_f32_16x16x32_bf16(false, (A), false, (B), (short)0, (C), false, false)

// A-fragment: 16x32 bf16, row-major LDS [row][k], stride even.
// lane<16: M=lane ; lane>=16 same M, upper K-half per ISA table.
__device__ __forceinline__ v16bf load_a(const __bf16* base, int stride,
                                        int rowBase, int kb, int lane) {
  const unsigned int* p = (const unsigned int*)base;
  int row  = rowBase + (lane & 15);
  int half = lane >> 4;
  FragU r;
#pragma unroll
  for (int j = 0; j < 8; ++j) {
    int kk = kb * 32 + ((j & 4) << 2) + half * 8 + ((j & 3) << 1);
    r.u[j] = p[(row * stride + kk) >> 1];
  }
  return r.b;
}

// B-fragment: 32x16 bf16, B[k][o] = W[o][k] with W row-major LDS [o][k].
// col = lane&15 ; VGPR j holds K = (j&4)*4 + half*8 + 2*(j&3), +1.
__device__ __forceinline__ v16bf load_b(const __bf16* base, int stride,
                                        int oBase, int kb, int lane) {
  const unsigned int* p = (const unsigned int*)base;
  int o    = oBase + (lane & 15);
  int half = lane >> 4;
  FragU r;
#pragma unroll
  for (int j = 0; j < 8; ++j) {
    int kk = kb * 32 + ((j & 4) << 2) + half * 8 + ((j & 3) << 1);
    r.u[j] = p[(o * stride + kk) >> 1];
  }
  return r.b;
}

__global__ __launch_bounds__(NTHREADS)
void gnn_gru_fused(const float* __restrict__ x,
                   const float* __restrict__ edge_attr,
                   const float* __restrict__ hidden,
                   const float* __restrict__ Wl, const float* __restrict__ bl,
                   const float* __restrict__ Wr, const float* __restrict__ br,
                   const float* __restrict__ We, const float* __restrict__ be,
                   const float* __restrict__ att, const float* __restrict__ bias_g,
                   const float* __restrict__ W_ih, const float* __restrict__ W_hh,
                   const float* __restrict__ b_ih, const float* __restrict__ b_hh,
                   const float* __restrict__ ln_g, const float* __restrict__ ln_b,
                   const float* __restrict__ Wq,  const float* __restrict__ bq,
                   float* __restrict__ out_q, float* __restrict__ out_h)
{
  extern __shared__ char smem[];
  __bf16* sW    = (__bf16*)(smem + OFF_W);
  __bf16* sX    = (__bf16*)(smem + OFF_X);
  float*  sXLR  = (float*)(smem + OFF_XLR);
  __bf16* sHH   = (__bf16*)(smem + OFF_HH);
  float*  sAGG  = (float*)(smem + OFF_AGG);
  float*  sHOUT = (float*)(smem + OFF_HOUT);
  float*  sSTAT = (float*)(smem + OFF_STAT);
  float*  sGG   = (float*)(smem + OFF_XLR);   // alias (stage-3 output)
  float*  sXF   = (float*)(smem + OFF_AGG);   // alias (stage-0 fp32 x staging)

  const int tid   = threadIdx.x;
  const int lane  = tid & 31;
  const int wv    = tid >> 5;
  const int gbase = blockIdx.x * NODES;       // global node base
  const int epg0  = blockIdx.x * NEPW;        // global episode base

  // -------- stage 0a: prefetch weights, async-stage x tile --------
#if defined(__gfx1250__)
  {
    // hint the weight matrices into cache (global_prefetch_b8)
    __builtin_prefetch(Wl + tid * 128, 0, 0);
    __builtin_prefetch(Wr + tid * 128, 0, 0);
    __builtin_prefetch(W_ih + tid * 96, 0, 0);
    __builtin_prefetch(W_hh + tid * 96, 0, 0);
    // async copy of x tile (32 x 64 fp32 = 8KB) into LDS, tracked by ASYNCcnt
    unsigned ldsBase =
        (unsigned)(size_t)(__attribute__((address_space(3))) char*)(void*)sXF;
    unsigned long long gaddr = (unsigned long long)(x + (size_t)gbase * 64);
#pragma unroll
    for (int it = 0; it < 4; ++it) {          // 128 lanes x 16B x 4 = 8KB
      unsigned voff = (unsigned)(tid + it * NTHREADS) * 16u;
      unsigned vdst = ldsBase + voff;
      asm volatile("global_load_async_to_lds_b128 %0, %1, %2"
                   :: "v"(vdst), "v"(voff), "s"(gaddr) : "memory");
    }
    asm volatile("s_wait_asynccnt 0x0" ::: "memory");
  }
  __syncthreads();
  for (int i = tid; i < NODES * 64; i += NTHREADS) {
    int nd = i >> 6, k = i & 63;
    sX[nd * 68 + k] = (__bf16)sXF[i];
  }
#else
  for (int i = tid; i < NODES * 64; i += NTHREADS) {
    int nd = i >> 6, k = i & 63;
    sX[nd * 68 + k] = (__bf16)x[(gbase + nd) * 64 + k];
  }
#endif
  // -------- stage 0b: stage [Wl;Wr] (512x64) as bf16 --------
  for (int i = tid; i < 512 * 64; i += NTHREADS) {
    int o = i >> 6, k = i & 63;
    float w = (o < 256) ? Wl[o * 64 + k] : Wr[(o - 256) * 64 + k];
    sW[o * 72 + k] = (__bf16)w;
  }
  __syncthreads();

  // -------- stage 1: xl|xr = X @ [Wl;Wr]^T  (WMMA bf16, K=64) --------
  // wave owns one M-tile + 16 N-tiles: A fragments loaded once, reused 16x
  {
    const int mt     = wv & 1;
    const int otBase = (wv >> 1) * 16;
    v16bf a0 = load_a(sX, 68, mt * 16, 0, lane);
    v16bf a1 = load_a(sX, 68, mt * 16, 1, lane);
    const int rb = mt * 16 + (lane >> 4) * 8;
    for (int ot = otBase; ot < otBase + 16; ++ot) {
      v16bf b0 = load_b(sW, 72, ot * 16, 0, lane);
      v16bf b1 = load_b(sW, 72, ot * 16, 1, lane);
      v8f c = {};
      c = WMMA_BF16(a0, b0, c);
      c = WMMA_BF16(a1, b1, c);
      int col  = ot * 16 + (lane & 15);
      float bi = (col < 256) ? bl[col] : br[col - 256];
#pragma unroll
      for (int r = 0; r < 8; ++r)
        sXLR[(rb + r) * 520 + col] = c[r] + bi;
    }
  }
  __syncthreads();

  // -------- stage 2: GATv2 attention, thread = (episode, dst, head) --------
  {
    const int epl = tid >> 5;        // local episode 0..3
    const int dh  = tid & 31;
    const int d   = dh >> 2;         // dst agent 0..7
    const int h   = dh & 3;          // head 0..3
    const int nl  = epl * 8 + d;     // local node
    const int epg = epg0 + epl;

    float ea0[7], ea1[7], ea2[7];
    int   sl[7];
#pragma unroll
    for (int si = 0; si < 7; ++si) {
      int s  = si + (si >= d ? 1 : 0);
      sl[si] = epl * 8 + s;
      int epos = s * 7 + ((d > s) ? d - 1 : d);    // edge position in episode
      const float* eap = edge_attr + (epg * 56 + epos) * 3;
      ea0[si] = eap[0]; ea1[si] = eap[1]; ea2[si] = eap[2];
    }
    float lg[7] = {0.f, 0.f, 0.f, 0.f, 0.f, 0.f, 0.f};
    for (int k = 0; k < 64; ++k) {
      int col    = h * 64 + k;
      float w0   = We[col * 3 + 0], w1 = We[col * 3 + 1], w2 = We[col * 3 + 2];
      float bek  = be[col];
      float xrk  = sXLR[nl * 520 + 256 + col];
      float attk = att[col];
#pragma unroll
      for (int si = 0; si < 7; ++si) {
        float m = sXLR[sl[si] * 520 + col] + xrk
                + (bek + w0 * ea0[si] + w1 * ea1[si] + w2 * ea2[si]);
        m = (m > 0.f) ? m : 0.2f * m;            // leaky relu 0.2
        lg[si] += attk * m;
      }
    }
    float mx = lg[0];
#pragma unroll
    for (int si = 1; si < 7; ++si) mx = fmaxf(mx, lg[si]);
    float al[7]; float sum = 0.f;
#pragma unroll
    for (int si = 0; si < 7; ++si) { al[si] = __expf(lg[si] - mx); sum += al[si]; }
    float inv = 1.f / sum;
    for (int k = 0; k < 64; ++k) {
      int col = h * 64 + k;
      float acc = 0.f;
#pragma unroll
      for (int si = 0; si < 7; ++si) acc += al[si] * sXLR[sl[si] * 520 + col];
      sAGG[nl * 256 + col] = acc * inv;
    }
  }
  __syncthreads();

  // -------- head mean + ReLU -> h_gnn bf16 ; stage hidden bf16 ; GRU weights --------
  for (int i = tid; i < NODES * 64; i += NTHREADS) {
    int nd = i >> 6, k = i & 63;
    const float* ag = sAGG + nd * 256;
    float v = 0.25f * (ag[k] + ag[64 + k] + ag[128 + k] + ag[192 + k]) + bias_g[k];
    v = fmaxf(v, 0.f);
    sHH[nd * 136 + k]      = (__bf16)v;
    sHH[nd * 136 + 64 + k] = (__bf16)hidden[(gbase + nd) * 64 + k];
  }
  for (int i = tid; i < 384 * 64; i += NTHREADS) {   // reuse weight LDS
    int o = i >> 6, k = i & 63;
    float w = (o < 192) ? W_ih[o * 64 + k] : W_hh[(o - 192) * 64 + k];
    sW[o * 72 + k] = (__bf16)w;
  }
  __syncthreads();

  // -------- stage 3: gi = h_gnn @ W_ih^T ; gh = hidden @ W_hh^T (WMMA) --------
  // wave owns (gemm, M-tile); A fragments hoisted over the 12 N-tiles
  {
    const int g  = wv & 1;                      // 0 -> gi, 1 -> gh
    const int mt = wv >> 1;
    v16bf a0 = load_a(sHH + g * 64, 136, mt * 16, 0, lane);
    v16bf a1 = load_a(sHH + g * 64, 136, mt * 16, 1, lane);
    const int rb = mt * 16 + (lane >> 4) * 8;
    for (int ot = 0; ot < 12; ++ot) {
      v16bf b0 = load_b(sW + g * 192 * 72, 72, ot * 16, 0, lane);
      v16bf b1 = load_b(sW + g * 192 * 72, 72, ot * 16, 1, lane);
      v8f c = {};
      c = WMMA_BF16(a0, b0, c);
      c = WMMA_BF16(a1, b1, c);
      int col = ot * 16 + (lane & 15);
#pragma unroll
      for (int r = 0; r < 8; ++r)
        sGG[(rb + r) * 400 + g * 200 + col] = c[r];
    }
  }
  __syncthreads();

  // -------- stage 4: GRU gates, h output --------
  for (int i = tid; i < NODES * 64; i += NTHREADS) {
    int nd = i >> 6, k = i & 63;
    const float* gg = sGG + nd * 400;
    float ir = gg[k]        + b_ih[k];
    float iz = gg[64 + k]   + b_ih[64 + k];
    float in_= gg[128 + k]  + b_ih[128 + k];
    float hr = gg[200 + k]  + b_hh[k];
    float hz = gg[264 + k]  + b_hh[64 + k];
    float hn = gg[328 + k]  + b_hh[128 + k];
    float r  = 1.f / (1.f + __expf(-(ir + hr)));
    float z  = 1.f / (1.f + __expf(-(iz + hz)));
    float n  = tanhf(in_ + r * hn);
    float hp = hidden[(gbase + nd) * 64 + k];
    float hv = (1.f - z) * n + z * hp;
    out_h[(gbase + nd) * 64 + k] = hv;
    sHOUT[nd * 64 + k] = hv;
  }
  __syncthreads();

  // -------- stage 5: LayerNorm stats --------
  if (tid < NODES) {
    float mu = 0.f;
    for (int k = 0; k < 64; ++k) mu += sHOUT[tid * 64 + k];
    mu *= (1.f / 64.f);
    float var = 0.f;
    for (int k = 0; k < 64; ++k) { float dv = sHOUT[tid * 64 + k] - mu; var += dv * dv; }
    var *= (1.f / 64.f);
    sSTAT[tid * 2]     = mu;
    sSTAT[tid * 2 + 1] = rsqrtf(var + 1e-5f);
  }
  __syncthreads();

  // -------- stage 6: q head (N x 10 GEMV) --------
  for (int i = tid; i < NODES * 10; i += NTHREADS) {
    int nd = i / 10, a = i % 10;
    float mu = sSTAT[nd * 2], rs = sSTAT[nd * 2 + 1];
    float acc = bq[a];
    for (int k = 0; k < 64; ++k) {
      float hln = (sHOUT[nd * 64 + k] - mu) * rs * ln_g[k] + ln_b[k];
      acc += hln * Wq[a * 64 + k];
    }
    out_q[(gbase + nd) * 10 + a] = acc;
  }
}

extern "C" void kernel_launch(void* const* d_in, const int* in_sizes, int n_in,
                              void* d_out, int out_size, void* d_ws, size_t ws_size,
                              hipStream_t stream) {
  (void)in_sizes; (void)n_in; (void)out_size; (void)d_ws; (void)ws_size;
  const float* x   = (const float*)d_in[0];
  const float* ea  = (const float*)d_in[1];
  const float* hid = (const float*)d_in[2];
  // d_in[3] = edge_index: structure is implicit (fully-connected groups of 8)
  const float* Wl  = (const float*)d_in[4];
  const float* bl  = (const float*)d_in[5];
  const float* Wr  = (const float*)d_in[6];
  const float* br  = (const float*)d_in[7];
  const float* We  = (const float*)d_in[8];
  const float* be  = (const float*)d_in[9];
  const float* att = (const float*)d_in[10];
  const float* bg  = (const float*)d_in[11];
  const float* Wih = (const float*)d_in[12];
  const float* Whh = (const float*)d_in[13];
  const float* bih = (const float*)d_in[14];
  const float* bhh = (const float*)d_in[15];
  const float* lng = (const float*)d_in[16];
  const float* lnb = (const float*)d_in[17];
  const float* Wq  = (const float*)d_in[18];
  const float* bq  = (const float*)d_in[19];

  float* out_q = (float*)d_out;
  float* out_h = out_q + (size_t)N_NODES * 10;

  (void)hipFuncSetAttribute((const void*)gnn_gru_fused,
                            hipFuncAttributeMaxDynamicSharedMemorySize,
                            SMEM_BYTES);
  dim3 grid(B_EP / NEPW);     // 1024 blocks
  dim3 block(NTHREADS);       // 4 waves
  gnn_gru_fused<<<grid, block, SMEM_BYTES, stream>>>(
      x, ea, hid, Wl, bl, Wr, br, We, be, att, bg,
      Wih, Whh, bih, bhh, lng, lnb, Wq, bq, out_q, out_h);
}
